// MonotonicNN_35167192220449
// MI455X (gfx1250) — matrix operand order repaired and checked by reading, not compiled, not run on hardware
//
#include <hip/hip_runtime.h>

#define S_STEPS 51
#define NB      50
#define IN_D    8
#define HID     128
#define NSAMP   32768
#define PI_F    3.14159265358979323846f

typedef __attribute__((ext_vector_type(16))) _Float16 v16h;
typedef __attribute__((ext_vector_type(2)))  _Float16 h2;
typedef __attribute__((ext_vector_type(8)))  float    v8f;

// B fragment (32x16, f16) from transposed weights Wt[col][k] (stride halves).
// Per lane half h: halves 0..7 <-> K = 8h + i, halves 8..15 <-> K = 16 + 8h + (i-8).
__device__ __forceinline__ v16h lds_b_frag(const _Float16* Wt, int stride, int ntile,
                                           int kc, int lane) {
  int col = ntile * 16 + (lane & 15);
  int kb  = kc * 32 + ((lane >> 4) << 3);
  const _Float16* p = Wt + col * stride + kb;
  v16h b;
#pragma unroll
  for (int i = 0; i < 8; ++i) { b[i] = p[i]; b[i + 8] = p[16 + i]; }
  return b;
}

// A fragment (16x32, f16) from row-major activations act[m][k] (stride HID).
__device__ __forceinline__ v16h lds_a_frag(const _Float16* act, int kc, int lane) {
  int m  = lane & 15;
  int kb = kc * 32 + ((lane >> 4) << 3);
  const _Float16* p = act + m * HID + kb;
  v16h a;
#pragma unroll
  for (int i = 0; i < 8; ++i) { a[i] = p[i]; a[i + 8] = p[16 + i]; }
  return a;
}

// ------- Kernel 1: integral MLP. Layer 0 collapsed to per-step affine map -----
// z0(col; s) = relu(c_s * u(col) + v(col)),  u,v step-invariant per sample.
// Per pair-iteration only layer-1 runs on WMMA (M=32: two steps share each B).
__global__ __launch_bounds__(128) void mono_integral_kernel(
    const float* __restrict__ xin,
    const float* __restrict__ iW0, const float* __restrict__ ib0,
    const float* __restrict__ iW1, const float* __restrict__ ib1,
    const float* __restrict__ iW2, const float* __restrict__ ib2,
    float* __restrict__ zs) {
  __shared__ _Float16 W1T[HID * HID];      // 32 KB, transposed f16
  __shared__ float    iW0s[IN_D * HID];    // 4 KB, f32 (row-major, as in memory)
  __shared__ float    ib0s[HID];
  __shared__ float    ccs[S_STEPS], css[S_STEPS];

  const int tid = threadIdx.x, lane = tid & 31, wid = tid >> 5;

  for (int i = tid; i < HID * HID; i += blockDim.x) {
    int c = i >> 7, k = i & 127;
    W1T[c * HID + k] = (_Float16)iW1[k * HID + c];
  }
  for (int i = tid; i < IN_D * HID; i += blockDim.x) iW0s[i] = iW0[i];
  for (int i = tid; i < HID; i += blockDim.x)        ib0s[i] = ib0[i];
  // Clenshaw-Curtis weights + step abscissae (exact integer range reduction)
  for (int j = tid; j < S_STEPS; j += blockDim.x) {
    float a = 0.f;
    for (int i = 0; i <= NB; i += 2) {
      float wi = (i == 0) ? 1.f : 2.f / (1.f - (float)(i * i));
      float l;
      if (j == 0) l = 0.5f;
      else {
        int ij = (i * j) % 100;
        l = __cosf((float)ij * (PI_F / (float)NB));
        if (j == NB) l *= 0.5f;
      }
      a += wi * l;
    }
    ccs[j] = a * (2.f / (float)NB);
    css[j] = (__cosf((float)j * PI_F / (float)NB) + 1.f) * 0.5f;
  }
  __syncthreads();

  const int base = (blockIdx.x * 4 + wid) * 16;
  const int row  = lane & 15;
  float xval = xin[(base + row) * IN_D + 0];
  float hfl[7];
#pragma unroll
  for (int j = 0; j < 7; ++j) hfl[j] = xin[(base + row) * IN_D + 1 + j];

  float ib1v[8], w2v[8];
#pragma unroll
  for (int n = 0; n < 8; ++n) {
    int col = n * 16 + (lane & 15);
    ib1v[n] = ib1[col];
    w2v[n]  = iW2[col];
  }
  const float ib2s = ib2[0];

  // u,v in layer-1 A-fragment order, packed f16 pairs (K-consecutive pairs).
  // half i of chunk kc maps to col = kc*32 + (i/8)*16 + (lane/16)*8 + (i%8)
  h2 upk[4][8], vpk[4][8];
#pragma unroll
  for (int kc = 0; kc < 4; ++kc) {
#pragma unroll
    for (int p = 0; p < 8; ++p) {
      float uu[2], vv[2];
#pragma unroll
      for (int e = 0; e < 2; ++e) {
        int i   = 2 * p + e;
        int col = kc * 32 + ((i >> 3) << 4) + ((lane >> 4) << 3) + (i & 7);
        uu[e] = xval * iW0s[col];            // iW0 row 0
        float v = ib0s[col];
#pragma unroll
        for (int j = 0; j < 7; ++j) v += hfl[j] * iW0s[(j + 1) * HID + col];
        vv[e] = v;
      }
      upk[kc][p][0] = (_Float16)uu[0]; upk[kc][p][1] = (_Float16)uu[1];
      vpk[kc][p][0] = (_Float16)vv[0]; vpk[kc][p][1] = (_Float16)vv[1];
    }
  }

  float accum[8];
#pragma unroll
  for (int r = 0; r < 8; ++r) accum[r] = 0.f;
  const int mrow = (lane >> 4) * 8;
  const _Float16 hz = (_Float16)0.f;

#pragma unroll 1
  for (int sp = 0; sp < 26; ++sp) {
    const int s0 = 2 * sp;
    int s1 = s0 + 1;
    float cw0 = ccs[s0], cw1;
    if (s1 > NB) { s1 = NB; cw1 = 0.f; }   // pad step, weight 0: uniform WMMA path
    else           cw1 = ccs[s1];
    const _Float16 c0h = (_Float16)css[s0];
    const _Float16 c1h = (_Float16)css[s1];
    h2 cs0; cs0[0] = c0h; cs0[1] = c0h;
    h2 cs1; cs1[0] = c1h; cs1[1] = c1h;

    // Build both step-tiles' A fragments in registers: relu(cs*u + v)
    v16h a1A[4], a1B[4];
#pragma unroll
    for (int kc = 0; kc < 4; ++kc) {
#pragma unroll
      for (int p = 0; p < 8; ++p) {
        h2 t0 = upk[kc][p] * cs0 + vpk[kc][p];   // v_pk_fma_f16
        h2 t1 = upk[kc][p] * cs1 + vpk[kc][p];
        a1A[kc][2 * p]     = (t0[0] > hz) ? t0[0] : hz;
        a1A[kc][2 * p + 1] = (t0[1] > hz) ? t0[1] : hz;
        a1B[kc][2 * p]     = (t1[0] > hz) ? t1[0] : hz;
        a1B[kc][2 * p + 1] = (t1[1] > hz) ? t1[1] : hz;
      }
    }

    // Layer 1 (128x128) on both step-tiles + fused 128->1 head dot
    float pA[8], pB[8];
#pragma unroll
    for (int r = 0; r < 8; ++r) { pA[r] = 0.f; pB[r] = 0.f; }
#pragma unroll
    for (int n = 0; n < 8; ++n) {
      v8f cA, cB;
#pragma unroll
      for (int r = 0; r < 8; ++r) { cA[r] = ib1v[n]; cB[r] = ib1v[n]; }
#pragma unroll
      for (int kc = 0; kc < 4; ++kc) {
        v16h b = lds_b_frag(W1T, HID, n, kc, lane);   // loaded once, used twice
        cA = __builtin_amdgcn_wmma_f32_16x16x32_f16(false, a1A[kc], false, b,
                                                    (short)0, cA, false, false);
        cB = __builtin_amdgcn_wmma_f32_16x16x32_f16(false, a1B[kc], false, b,
                                                    (short)0, cB, false, false);
      }
#pragma unroll
      for (int r = 0; r < 8; ++r) {
        pA[r] += fmaxf(cA[r], 0.f) * w2v[n];
        pB[r] += fmaxf(cB[r], 0.f) * w2v[n];
      }
    }
    // butterfly over 16-lane halves: rows mrow..mrow+7, both steps
#pragma unroll
    for (int r = 0; r < 8; ++r) {
      float p = pA[r], q = pB[r];
      p += __shfl_xor(p, 1, 32);  q += __shfl_xor(q, 1, 32);
      p += __shfl_xor(p, 2, 32);  q += __shfl_xor(q, 2, 32);
      p += __shfl_xor(p, 4, 32);  q += __shfl_xor(q, 4, 32);
      p += __shfl_xor(p, 8, 32);  q += __shfl_xor(q, 8, 32);
      float yA = p + ib2s, yB = q + ib2s;
      float fA = (yA > 0.f) ? (yA + 1.f) : __expf(yA);   // elu(y)+1
      float fB = (yB > 0.f) ? (yB + 1.f) : __expf(yB);
      accum[r] += fA * cw0 + fB * cw1;
    }
  }

  if ((lane & 15) == 0) {
#pragma unroll
    for (int r = 0; r < 8; ++r) zs[base + mrow + r] = accum[r];
  }
}

// ---------------- Kernel 2: head MLP (7->128->128->2) + sigmoid ----------------
__global__ __launch_bounds__(128) void mono_head_kernel(
    const float* __restrict__ xin,
    const float* __restrict__ nW0, const float* __restrict__ nb0,
    const float* __restrict__ nW1, const float* __restrict__ nb1,
    const float* __restrict__ nW2, const float* __restrict__ nb2,
    const float* __restrict__ zs, float* __restrict__ out) {
  __shared__ _Float16 N0T[HID * 32];
  __shared__ _Float16 N1T[HID * HID];
  __shared__ _Float16 Act[4][16 * HID];

  const int tid = threadIdx.x, lane = tid & 31, wid = tid >> 5;
  for (int i = tid; i < HID * 32; i += blockDim.x) {
    int c = i >> 5, k = i & 31;
    N0T[c * 32 + k] = (k < IN_D - 1) ? (_Float16)nW0[k * HID + c] : (_Float16)0.f;
  }
  for (int i = tid; i < HID * HID; i += blockDim.x) {
    int c = i >> 7, k = i & 127;
    N1T[c * HID + k] = (_Float16)nW1[k * HID + c];
  }
  __syncthreads();

  const int base = (blockIdx.x * 4 + wid) * 16;
  const int row  = lane & 15;
  _Float16 hf[7];
#pragma unroll
  for (int j = 0; j < 7; ++j) hf[j] = (_Float16)xin[(base + row) * IN_D + 1 + j];

  float nb0v[8], nb1v[8], w2a[8], w2b[8];
#pragma unroll
  for (int n = 0; n < 8; ++n) {
    int col = n * 16 + (lane & 15);
    nb0v[n] = nb0[col];
    nb1v[n] = nb1[col];
    w2a[n]  = nW2[col * 2 + 0];
    w2b[n]  = nW2[col * 2 + 1];
  }
  const float b2a = nb2[0], b2b = nb2[1];
  _Float16* act = Act[wid];

  v16h a0;
#pragma unroll
  for (int i = 0; i < 16; ++i) a0[i] = (_Float16)0.f;
  if (lane < 16) {
#pragma unroll
    for (int j = 0; j < 7; ++j) a0[j] = hf[j];
  }

  const int mrow = (lane >> 4) * 8;
#pragma unroll
  for (int n = 0; n < 8; ++n) {
    v8f c;
#pragma unroll
    for (int r = 0; r < 8; ++r) c[r] = nb0v[n];
    v16h b = lds_b_frag(N0T, 32, n, 0, lane);
    c = __builtin_amdgcn_wmma_f32_16x16x32_f16(false, a0, false, b,
                                               (short)0, c, false, false);
    int colb = n * 16 + (lane & 15);
#pragma unroll
    for (int r = 0; r < 8; ++r)
      act[(mrow + r) * HID + colb] = (_Float16)fmaxf(c[r], 0.f);
  }

  v16h a1[4];
#pragma unroll
  for (int kc = 0; kc < 4; ++kc) a1[kc] = lds_a_frag(act, kc, lane);
  float pa[8], pb[8];
#pragma unroll
  for (int r = 0; r < 8; ++r) { pa[r] = 0.f; pb[r] = 0.f; }
#pragma unroll
  for (int n = 0; n < 8; ++n) {
    v8f c;
#pragma unroll
    for (int r = 0; r < 8; ++r) c[r] = nb1v[n];
#pragma unroll
    for (int kc = 0; kc < 4; ++kc) {
      v16h b = lds_b_frag(N1T, HID, n, kc, lane);
      c = __builtin_amdgcn_wmma_f32_16x16x32_f16(false, a1[kc], false, b,
                                                 (short)0, c, false, false);
    }
#pragma unroll
    for (int r = 0; r < 8; ++r) {
      float z = fmaxf(c[r], 0.f);
      pa[r] += z * w2a[n];
      pb[r] += z * w2b[n];
    }
  }
#pragma unroll
  for (int r = 0; r < 8; ++r) {
    float a = pa[r], b = pb[r];
    a += __shfl_xor(a, 1, 32); a += __shfl_xor(a, 2, 32);
    a += __shfl_xor(a, 4, 32); a += __shfl_xor(a, 8, 32);
    b += __shfl_xor(b, 1, 32); b += __shfl_xor(b, 2, 32);
    b += __shfl_xor(b, 4, 32); b += __shfl_xor(b, 8, 32);
    if ((lane & 15) == 0) {
      int rr = base + mrow + r;
      float offset  = a + b2a;
      float scaling = __expf(b + b2b);
      float integ   = zs[rr] * xin[rr * IN_D] * 0.5f;
      float v = scaling * integ + offset;
      out[rr] = 1.f / (1.f + __expf(-v));
    }
  }
}

extern "C" void kernel_launch(void* const* d_in, const int* in_sizes, int n_in,
                              void* d_out, int out_size, void* d_ws, size_t ws_size,
                              hipStream_t stream) {
  (void)in_sizes; (void)n_in; (void)out_size; (void)ws_size;
  const float* xin = (const float*)d_in[0];
  const float* iW0 = (const float*)d_in[1];  const float* ib0 = (const float*)d_in[2];
  const float* iW1 = (const float*)d_in[3];  const float* ib1 = (const float*)d_in[4];
  const float* iW2 = (const float*)d_in[5];  const float* ib2 = (const float*)d_in[6];
  const float* nW0 = (const float*)d_in[7];  const float* nb0 = (const float*)d_in[8];
  const float* nW1 = (const float*)d_in[9];  const float* nb1 = (const float*)d_in[10];
  const float* nW2 = (const float*)d_in[11]; const float* nb2 = (const float*)d_in[12];
  float* zs  = (float*)d_ws;   // N floats of per-sample z_est
  float* out = (float*)d_out;

  dim3 grid1(NSAMP / 64), block1(128);   // 4 waves/WG, 16 samples/wave, 2 steps/iter
  hipLaunchKernelGGL(mono_integral_kernel, grid1, block1, 0, stream,
                     xin, iW0, ib0, iW1, ib1, iW2, ib2, zs);
  dim3 grid2(NSAMP / 64), block2(128);   // 4 waves/WG, 16 samples/wave
  hipLaunchKernelGGL(mono_head_kernel, grid2, block2, 0, stream,
                     xin, nW0, nb0, nW1, nb1, nW2, nb2, zs, out);
}
